// MarlinLinear_33260226740741
// MI455X (gfx1250) — matrix operand order repaired and checked by reading, not compiled, not run on hardware
//
#include <hip/hip_runtime.h>

typedef __attribute__((ext_vector_type(16))) _Float16 v16h;
typedef __attribute__((ext_vector_type(8)))  _Float16 h8;
typedef __attribute__((ext_vector_type(4)))  _Float16 h4;
typedef __attribute__((ext_vector_type(8)))  float    v8f;
typedef __attribute__((ext_vector_type(4)))  float    f4;
typedef __attribute__((ext_vector_type(4)))  int      i4;
typedef __attribute__((ext_vector_type(4)))  unsigned int u32x4;
typedef __attribute__((ext_vector_type(8)))  int      i32x8;
typedef __attribute__((ext_vector_type(4)))  int      i32x4;

#define R_TOT  8192          // 4*2048 rows of x
#define K_TOT  4096
#define N_TOT  4096
#define NGROUP 32            // K_TOT / 128
#define RT     128           // block tile rows
#define NT     128           // block tile cols
#define KTILE  32            // K step (one WMMA K per step)
#define NKT    (K_TOT / KTILE)
#define LSTR   40            // LDS stride in halves (32 data + 8 pad; 80B, 16B-aligned)

#if __has_builtin(__builtin_amdgcn_tensor_load_to_lds)
#define HAVE_TDM 1
#else
#define HAVE_TDM 0
#endif

__device__ __forceinline__ v16h cat16(h8 lo, h8 hi) {
  return __builtin_shufflevector(lo, hi, 0,1,2,3,4,5,6,7,8,9,10,11,12,13,14,15);
}

#if HAVE_TDM
// TDM descriptor (D#) per CDNA5 ISA:
// 2D tile: tile_dim0 = 32 elems (K, contiguous), tile_dim1 = 16 rows (N),
// data_size = 2B, tensor row stride = 4096 elems. LDS destination gets
// 4 dwords of pad after every 16 dwords -> 40-half rows matching LSTR.
__device__ __forceinline__ void tdm_load_w_tile(const _Float16* gsrc, unsigned lds_off) {
  unsigned long long ga = (unsigned long long)(uintptr_t)gsrc;
  u32x4 g0 = {};
  g0[0] = 1u;                                           // count = 1 (valid)
  g0[1] = lds_off;                                      // lds_addr (bytes)
  g0[2] = (unsigned)(ga & 0xFFFFFFFFu);                 // global_addr[31:0]
  g0[3] = (unsigned)((ga >> 32) & 0x01FFFFFFu)          // global_addr[56:32]
        | (2u << 30);                                   // type = 2 ("image")
  i32x8 g1 = {};
  g1[0] = (1 << 16)                                     // data_size = 2B
        | (1 << 20)                                     // pad_enable
        | (3 << 22)                                     // pad_interval: 16 dwords
        | (3 << 25);                                    // pad_amount: 4 dwords
  g1[1] = (int)((K_TOT & 0xFFFF) << 16);                // tensor_dim0 lo16
  g1[2] = (int)(((K_TOT >> 16) & 0xFFFF)                // tensor_dim0 hi16
        | ((N_TOT & 0xFFFF) << 16));                    // tensor_dim1 lo16
  g1[3] = (int)(((N_TOT >> 16) & 0xFFFF)                // tensor_dim1 hi16
        | (KTILE << 16));                               // tile_dim0 = 32
  g1[4] = 16;                                           // tile_dim1 = 16, tile_dim2 = 0
  g1[5] = K_TOT;                                        // tensor_dim0_stride lo32
  g1[6] = (int)((K_TOT & 0xFFFF) << 16);                // stride0 hi=0 | stride1 lo16
  g1[7] = 0;                                            // stride1 hi
  i32x4 z4 = {};
#if defined(__clang_major__) && (__clang_major__ >= 23)
  i32x8 z8 = {};
  __builtin_amdgcn_tensor_load_to_lds(g0, g1, z4, z4, z8, 0);
#else
  __builtin_amdgcn_tensor_load_to_lds(g0, g1, z4, z4, 0);
#endif
}
#endif

// ---------------------------------------------------------------------------
// Phase 1: one-shot dequant Wq[k][n] (int4-in-int32) -> Wh[n][k] f16, LDS transpose.
// Runs once per call; ~100 MB of traffic vs ~275 GFLOP GEMM -> negligible.
// ---------------------------------------------------------------------------
__global__ __launch_bounds__(256)
void marlin_dequant_kernel(const int* __restrict__ Wq,
                           const float* __restrict__ scales,
                           _Float16* __restrict__ Wh)
{
  __shared__ __align__(16) _Float16 t[128 * LSTR];
  const int tid = threadIdx.x;
  const int n0  = blockIdx.x * 128;
  const int k0  = blockIdx.y * 32;
  const int n4  = tid & 31;          // 4-col group in N
  const int kq  = tid >> 5;          // 0..7, each covers 4 k
  const int g   = k0 >> 7;           // 32-k chunk never crosses a group
  f4 sv = *(const f4*)&scales[g * N_TOT + n0 + n4 * 4];
  #pragma unroll
  for (int j = 0; j < 4; ++j) {
    const int k = kq * 4 + j;
    i4 q = *(const i4*)&Wq[(k0 + k) * N_TOT + n0 + n4 * 4];
    #pragma unroll
    for (int i = 0; i < 4; ++i)
      t[(n4 * 4 + i) * LSTR + k] = (_Float16)((float)(q[i] - 8) * sv[i]);
  }
  __syncthreads();
  #pragma unroll
  for (int c = 0; c < 2; ++c) {
    const int chunk = tid + 256 * c;       // 512 h8 chunks: 128 n x 4
    const int n = chunk >> 2, p = chunk & 3;
    *(h8*)&Wh[(size_t)(n0 + n) * K_TOT + k0 + p * 8] = *(const h8*)&t[n * LSTR + p * 8];
  }
}

// ---------------------------------------------------------------------------
// Phase 2: f16 GEMM. W tiles arrive in LDS via the Tensor Data Mover (one
// tensor_load_to_lds per wave per K-step); x tiles are converted fp32->f16 by
// VALU (irreducible). 8 waves, 128x128 block tile, 8 WMMA per wave per step.
// ---------------------------------------------------------------------------
__global__ __launch_bounds__(256)
void marlin_gemm_f16w_kernel(const float* __restrict__ x,
                             const _Float16* __restrict__ Wh,
                             const float* __restrict__ u,
                             const float* __restrict__ bias,
                             float* __restrict__ out)
{
  __shared__ __align__(16) _Float16 xs[2][RT * LSTR];   // x tile f16 [m][k]
  __shared__ __align__(16) _Float16 ws[2][NT * LSTR];   // W tile f16 [n][k]
  __shared__ float xsum[RT];

  const int tid  = threadIdx.x;
  const int n0   = blockIdx.x * NT;
  const int r0   = blockIdx.y * RT;
  const int lane = tid & 31;
  const int wv   = tid >> 5;
  const int wr   = wv & 1;
  const int wn   = wv >> 1;
  const int row  = lane & 15;
  const int kg   = lane >> 4;

  if (tid < RT) xsum[tid] = 0.0f;

  v8f acc[4][2];
  #pragma unroll
  for (int s = 0; s < 4; ++s)
    #pragma unroll
    for (int t = 0; t < 2; ++t)
      acc[s][t] = {};

  float xpart[4] = {0.f, 0.f, 0.f, 0.f};
  f4 xv[4];

  auto loadX = [&](int kt) {
    const int k0 = kt * KTILE;
    #pragma unroll
    for (int r = 0; r < 4; ++r) {
      int idx = tid + 256 * r;
      int m = idx >> 3, k4 = idx & 7;
      xv[r] = *(const f4*)&x[(r0 + m) * K_TOT + k0 + k4 * 4];
      if (r == 0 && k0 + KTILE < K_TOT)
        __builtin_prefetch(&x[(r0 + m) * K_TOT + k0 + KTILE + k4 * 4], 0, 1);
    }
  };
  auto storeX = [&](int buf, int kt) {
    (void)kt;
    #pragma unroll
    for (int r = 0; r < 4; ++r) {
      int idx = tid + 256 * r;
      int m = idx >> 3, k4 = idx & 7;
      f4 v = xv[r];
      xpart[r] += v[0] + v[1] + v[2] + v[3];
      h4 xh;
      #pragma unroll
      for (int i = 0; i < 4; ++i) xh[i] = (_Float16)v[i];
      *(h4*)&xs[buf][m * LSTR + k4 * 4] = xh;
    }
  };
  auto loadW = [&](int buf, int kt) {       // async: TDM engine fills LDS
#if HAVE_TDM
    tdm_load_w_tile(Wh + (size_t)(n0 + wv * 16) * K_TOT + kt * KTILE,
                    (unsigned)(uintptr_t)&ws[buf][(wv * 16) * LSTR]);
#else
    (void)buf; (void)kt;
#endif
  };
  auto copyW = [&](int buf, int kt) {       // fallback if no TDM builtin
#if !HAVE_TDM
    #pragma unroll
    for (int c = 0; c < 2; ++c) {
      const int chunk = tid + 256 * c;
      const int n = chunk >> 2, p = chunk & 3;
      *(h8*)&ws[buf][n * LSTR + p * 8] =
          *(const h8*)&Wh[(size_t)(n0 + n) * K_TOT + kt * KTILE + p * 8];
    }
#else
    (void)buf; (void)kt;
#endif
  };

  // ---- pipeline prologue ----
  loadX(0);
  loadW(0, 0);
  storeX(0, 0);
  copyW(0, 0);
#if HAVE_TDM
  __builtin_amdgcn_s_wait_tensorcnt((short)0);
#endif
  __syncthreads();

  // ---- main K loop ----
  for (int kt = 0; kt < NKT; ++kt) {
    const int cur = kt & 1;
    if (kt + 1 < NKT) { loadX(kt + 1); loadW(1 - cur, kt + 1); }

    const _Float16* xb = xs[cur];
    const _Float16* wb = ws[cur];

    v16h a[4];
    #pragma unroll
    for (int s = 0; s < 4; ++s) {
      const _Float16* p = xb + (wr * 64 + s * 16 + row) * LSTR;
      h8 lo = *(const h8*)(p + kg * 8);
      h8 hi = *(const h8*)(p + 16 + kg * 8);
      a[s] = cat16(lo, hi);
    }
    v16h b[2];
    #pragma unroll
    for (int t = 0; t < 2; ++t) {
      const _Float16* p = wb + (wn * 32 + t * 16 + row) * LSTR;
      h8 lo = *(const h8*)(p + kg * 8);
      h8 hi = *(const h8*)(p + 16 + kg * 8);
      b[t] = cat16(lo, hi);
    }

    #pragma unroll
    for (int s = 0; s < 4; ++s)
      #pragma unroll
      for (int t = 0; t < 2; ++t)
        acc[s][t] = __builtin_amdgcn_wmma_f32_16x16x32_f16(
            false, a[s], false, b[t], (short)0, acc[s][t], false, false);

    if (kt + 1 < NKT) { storeX(1 - cur, kt + 1); copyW(1 - cur, kt + 1); }
#if HAVE_TDM
    __builtin_amdgcn_s_wait_tensorcnt((short)0);
#endif
    __syncthreads();
  }

  // ---- rank-1 row-sum reduction ----
  #pragma unroll
  for (int r = 0; r < 4; ++r)
    atomicAdd(&xsum[(tid + 256 * r) >> 3], xpart[r]);
  __syncthreads();

  // ---- epilogue ----
  #pragma unroll
  for (int t = 0; t < 2; ++t) {
    const int nloc = wn * 32 + t * 16 + row;
    const int n    = n0 + nloc;
    const float uv = u[n];
    const float bv = bias[n];
    #pragma unroll
    for (int s = 0; s < 4; ++s) {
      v8f c = acc[s][t];
      #pragma unroll
      for (int v = 0; v < 8; ++v) {
        const int rl = wr * 64 + s * 16 + kg * 8 + v;
        out[(r0 + rl) * N_TOT + n] = c[v] + xsum[rl] * uv + bv;
      }
    }
  }
}

// ---------------------------------------------------------------------------
// Fallback: round-1 single-pass fused kernel (used if d_ws is too small).
// ---------------------------------------------------------------------------
__global__ __launch_bounds__(256)
void marlin_fused_kernel(const float* __restrict__ x,
                         const int*   __restrict__ Wq,
                         const float* __restrict__ scales,
                         const float* __restrict__ u,
                         const float* __restrict__ bias,
                         float* __restrict__ out)
{
  __shared__ __align__(16) _Float16 xs[2][RT * LSTR];
  __shared__ __align__(16) _Float16 ws[2][NT * LSTR];
  __shared__ __align__(16) float    sc[NGROUP * NT];
  __shared__ float xsum[RT];

  const int tid  = threadIdx.x;
  const int n0   = blockIdx.x * NT;
  const int r0   = blockIdx.y * RT;
  const int lane = tid & 31;
  const int wv   = tid >> 5;
  const int wr   = wv & 1;
  const int wn   = wv >> 1;
  const int row  = lane & 15;
  const int kg   = lane >> 4;

  if (tid < RT) xsum[tid] = 0.0f;
  #pragma unroll
  for (int r = 0; r < 4; ++r) {
    int idx = tid + 256 * r;
    int g   = idx >> 5;
    int nl  = (idx * 4) & 127;
    *(f4*)&sc[idx * 4] = *(const f4*)&scales[g * N_TOT + n0 + nl];
  }

  v8f acc[4][2];
  #pragma unroll
  for (int s = 0; s < 4; ++s)
    #pragma unroll
    for (int t = 0; t < 2; ++t)
      acc[s][t] = {};

  float xpart[4] = {0.f, 0.f, 0.f, 0.f};
  f4 xv[4];
  i4 wq[4];

  auto loadTile = [&](int kt) {
    const int k0 = kt * KTILE;
    #pragma unroll
    for (int r = 0; r < 4; ++r) {
      int idx = tid + 256 * r;
      int m   = idx >> 3, k4 = idx & 7;
      xv[r] = *(const f4*)&x[(r0 + m) * K_TOT + k0 + k4 * 4];
      int kk  = idx >> 5, n4 = idx & 31;
      wq[r] = *(const i4*)&Wq[(k0 + kk) * N_TOT + n0 + n4 * 4];
    }
  };
  auto storeTile = [&](int buf, int kt) {
    const int g = (kt * KTILE) >> 7;
    #pragma unroll
    for (int r = 0; r < 4; ++r) {
      int idx = tid + 256 * r;
      int m   = idx >> 3, k4 = idx & 7;
      f4 v = xv[r];
      xpart[r] += v[0] + v[1] + v[2] + v[3];
      h4 xh;
      #pragma unroll
      for (int i = 0; i < 4; ++i) xh[i] = (_Float16)v[i];
      *(h4*)&xs[buf][m * LSTR + k4 * 4] = xh;
      int kk = idx >> 5, n4 = idx & 31;
      f4 sv = *(const f4*)&sc[g * NT + n4 * 4];
      i4 q  = wq[r];
      #pragma unroll
      for (int i = 0; i < 4; ++i)
        ws[buf][(n4 * 4 + i) * LSTR + kk] = (_Float16)((float)(q[i] - 8) * sv[i]);
    }
  };

  loadTile(0);
  __syncthreads();
  storeTile(0, 0);
  __syncthreads();

  for (int kt = 0; kt < NKT; ++kt) {
    const int cur = kt & 1;
    if (kt + 1 < NKT) loadTile(kt + 1);

    const _Float16* xb = xs[cur];
    const _Float16* wb = ws[cur];
    v16h a[4];
    #pragma unroll
    for (int s = 0; s < 4; ++s) {
      const _Float16* p = xb + (wr * 64 + s * 16 + row) * LSTR;
      h8 lo = *(const h8*)(p + kg * 8);
      h8 hi = *(const h8*)(p + 16 + kg * 8);
      a[s] = cat16(lo, hi);
    }
    v16h b[2];
    #pragma unroll
    for (int t = 0; t < 2; ++t) {
      const _Float16* p = wb + (wn * 32 + t * 16 + row) * LSTR;
      h8 lo = *(const h8*)(p + kg * 8);
      h8 hi = *(const h8*)(p + 16 + kg * 8);
      b[t] = cat16(lo, hi);
    }
    #pragma unroll
    for (int s = 0; s < 4; ++s)
      #pragma unroll
      for (int t = 0; t < 2; ++t)
        acc[s][t] = __builtin_amdgcn_wmma_f32_16x16x32_f16(
            false, a[s], false, b[t], (short)0, acc[s][t], false, false);

    if (kt + 1 < NKT) storeTile(1 - cur, kt + 1);
    __syncthreads();
  }

  #pragma unroll
  for (int r = 0; r < 4; ++r)
    atomicAdd(&xsum[(tid + 256 * r) >> 3], xpart[r]);
  __syncthreads();

  #pragma unroll
  for (int t = 0; t < 2; ++t) {
    const int nloc = wn * 32 + t * 16 + row;
    const int n    = n0 + nloc;
    const float uv = u[n];
    const float bv = bias[n];
    #pragma unroll
    for (int s = 0; s < 4; ++s) {
      v8f c = acc[s][t];
      #pragma unroll
      for (int v = 0; v < 8; ++v) {
        const int rl = wr * 64 + s * 16 + kg * 8 + v;
        out[(r0 + rl) * N_TOT + n] = c[v] + xsum[rl] * uv + bv;
      }
    }
  }
}

extern "C" void kernel_launch(void* const* d_in, const int* in_sizes, int n_in,
                              void* d_out, int out_size, void* d_ws, size_t ws_size,
                              hipStream_t stream) {
  const float* x      = (const float*)d_in[0];
  const int*   Wq     = (const int*)  d_in[1];
  const float* scales = (const float*)d_in[2];
  const float* u      = (const float*)d_in[3];
  const float* bias   = (const float*)d_in[4];
  float*       out    = (float*)d_out;

  const size_t need = (size_t)K_TOT * N_TOT * sizeof(_Float16);   // 32 MB
  if (ws_size >= need) {
    _Float16* Wh = (_Float16*)d_ws;
    marlin_dequant_kernel<<<dim3(N_TOT / 128, K_TOT / 32), dim3(256), 0, stream>>>(
        Wq, scales, Wh);
    marlin_gemm_f16w_kernel<<<dim3(N_TOT / NT, R_TOT / RT), dim3(256), 0, stream>>>(
        x, Wh, u, bias, out);
  } else {
    marlin_fused_kernel<<<dim3(N_TOT / NT, R_TOT / RT), dim3(256), 0, stream>>>(
        x, Wq, scales, u, bias, out);
  }
}
